// Attention_1348619731153
// MI455X (gfx1250) — matrix operand order repaired and checked by reading, not compiled, not run on hardware
//
#include <hip/hip_runtime.h>
#include <hip/hip_bf16.h>

typedef __bf16 bf16;
typedef __attribute__((ext_vector_type(16))) __bf16 v16bf;
typedef __attribute__((ext_vector_type(8)))  __bf16 v8bf;
typedef __attribute__((ext_vector_type(8)))  float  v8f;

// ---------------------------------------------------------------------------
// WMMA fragment loaders (contiguous k order: lane half h holds k 16h..16h+15, wave32)
// ---------------------------------------------------------------------------

// A matrix 16x32 bf16, src row-major [m][k], ld in elements.
// lanes 0-15: m=lane, elems 0-15 -> k 0-15
// lanes 16-31: m=lane-16, elems 0-15 -> k 16-31
__device__ __forceinline__ v16bf load_a_frag(const bf16* src, int ld, int lane) {
  int m    = lane & 15;
  int koff = (lane < 16) ? 0 : 16;
  union { v16bf v; v8bf h[2]; } u;
  u.h[0] = *(const v8bf*)(src + (size_t)m * ld + koff);
  u.h[1] = *(const v8bf*)(src + (size_t)m * ld + 8 + koff);
  return u.v;
}

// B matrix 32x16 bf16 from K-contiguous ("transposed") storage srcT[n][k].
// lanes 0-15: n=lane, k 0-15; lanes 16-31: n=lane-16, k 16-31 (contiguous 32B)
__device__ __forceinline__ v16bf load_b_frag_t(const bf16* srcT, int ld, int lane) {
  int n = lane & 15;
  int k = (lane < 16) ? 0 : 16;
  return *(const v16bf*)(srcT + (size_t)n * ld + k);
}

#define WMMA_BF16(A, B, C) \
  __builtin_amdgcn_wmma_f32_16x16x32_bf16(false, (A), false, (B), (short)0, (C), false, false)

// ---------------------------------------------------------------------------
// f32 -> bf16 conversion (sizes multiples of 1024)
// ---------------------------------------------------------------------------
__global__ __launch_bounds__(256) void k_f32_to_bf16(const float* __restrict__ in,
                                                     bf16* __restrict__ out, int n) {
  int i = (blockIdx.x * 256 + threadIdx.x) * 4;
  if (i + 3 < n) {
    float4 f = *(const float4*)(in + i);
    out[i + 0] = (bf16)f.x;
    out[i + 1] = (bf16)f.y;
    out[i + 2] = (bf16)f.z;
    out[i + 3] = (bf16)f.w;
  }
}

// ---------------------------------------------------------------------------
// f32 [K][N] -> bf16 transposed [N][K], 32x32 LDS tiles, both sides coalesced
// ---------------------------------------------------------------------------
__global__ __launch_bounds__(256) void k_transpose_bf16(const float* __restrict__ in,
                                                        bf16* __restrict__ out,
                                                        int K, int N) {
  __shared__ float S[32][33];
  int kb = blockIdx.x * 32, nb = blockIdx.y * 32;
  int t = threadIdx.x;
  int cl = t & 31, rw = t >> 5;          // 8 row-groups of 32 lanes
#pragma unroll
  for (int i = 0; i < 4; ++i)
    S[rw + i * 8][cl] = in[(size_t)(kb + rw + i * 8) * N + nb + cl];
  __syncthreads();
#pragma unroll
  for (int i = 0; i < 4; ++i)
    out[(size_t)(nb + rw + i * 8) * K + kb + cl] = (bf16)S[cl][rw + i * 8];
}

// ---------------------------------------------------------------------------
// GEMM mainloop: block tile 128x64, 8 waves (4x2), each wave 32x32 (2x2 WMMA).
// A staged in LDS (double-buffered, one barrier per K-step); B-frags loaded
// straight from pre-transposed global weights Wt[n][k] (contiguous 32B/lane).
// ---------------------------------------------------------------------------
__device__ __forceinline__ void gemm_mainloop_t(const bf16* __restrict__ A,
                                                const bf16* __restrict__ Wt,
                                                int K, int mblk, int nblk,
                                                bf16* As /* 2*128*32 */,
                                                v8f acc[2][2],
                                                int tid, int lane, int wm, int wn) {
  int r = tid >> 1, c = (tid & 1) * 16;
  const bf16* Ap  = A + (size_t)(mblk + r) * K + c;
  const bf16* Bp0 = Wt + (size_t)(nblk + wn * 32) * K;
  const bf16* Bp1 = Wt + (size_t)(nblk + wn * 32 + 16) * K;

  // prime buffer 0
  *(v16bf*)(As + r * 32 + c) = *(const v16bf*)(Ap);

  int buf = 0;
  for (int k0 = 0; k0 < K; k0 += 32) {
    __syncthreads();
    if (k0 + 32 < K)
      *(v16bf*)(As + (buf ^ 1) * (128 * 32) + r * 32 + c) =
          *(const v16bf*)(Ap + k0 + 32);

    const bf16* Ab = As + buf * (128 * 32);
    v16bf af0 = load_a_frag(Ab + (wm * 32) * 32, 32, lane);
    v16bf af1 = load_a_frag(Ab + (wm * 32 + 16) * 32, 32, lane);
    v16bf bf0 = load_b_frag_t(Bp0 + k0, K, lane);
    v16bf bf1 = load_b_frag_t(Bp1 + k0, K, lane);
    acc[0][0] = WMMA_BF16(af0, bf0, acc[0][0]);
    acc[0][1] = WMMA_BF16(af0, bf1, acc[0][1]);
    acc[1][0] = WMMA_BF16(af1, bf0, acc[1][0]);
    acc[1][1] = WMMA_BF16(af1, bf1, acc[1][1]);
    buf ^= 1;
  }
}

// ---------------------------------------------------------------------------
// GEMM1: qkv = x @ w_qkv with fused RoPE epilogue.
// Writes bf16 q,k in [B,H,N,D] and V pre-transposed in [B,H,D,N].
// N-tile of 64 aligns exactly with one head of q|k|v (3072 = 48 heads * 64).
// ---------------------------------------------------------------------------
__global__ __launch_bounds__(256) void k_gemm_qkv_rope(const bf16* __restrict__ X,
                                                       const bf16* __restrict__ Wt,
                                                       bf16* __restrict__ qb,
                                                       bf16* __restrict__ kb,
                                                       bf16* __restrict__ vTb) {
  __shared__ alignas(32) unsigned char smem[34 * 1024];
  bf16*  As = (bf16*)smem;            // 2 x [128][32] = 16KB
  float* Cs = (float*)smem;           // [128][65] epilogue reuse (33.3KB)

  const int K = 1024;
  int tid = threadIdx.x, lane = tid & 31, wave = tid >> 5;
  int wm = wave & 3, wn = wave >> 2;
  int mblk = blockIdx.x * 128, nblk = blockIdx.y * 64;

  v8f acc[2][2] = {};
  gemm_mainloop_t(X, Wt, K, mblk, nblk, As, acc, tid, lane, wm, wn);

  __syncthreads();
  // spill accumulators (C layout: col = lane&15, row = v + 8*(lane>=16))
#pragma unroll
  for (int i = 0; i < 2; ++i)
#pragma unroll
    for (int j = 0; j < 2; ++j)
#pragma unroll
      for (int v = 0; v < 8; ++v) {
        int r = wm * 32 + i * 16 + v + ((lane >> 4) << 3);
        int c = wn * 32 + j * 16 + (lane & 15);
        Cs[r * 65 + c] = acc[i][j][v];
      }
  __syncthreads();

  int which = nblk / 1024;            // 0=q 1=k 2=v
  int h     = (nblk % 1024) / 64;     // head index

  if (which == 2) {
    // V: write transposed [B,H,D,N]; consecutive lanes -> consecutive n
    for (int t = tid; t < 128 * 64; t += 256) {
      int d = t >> 7, r = t & 127;
      int gm = mblk + r;
      int b = gm >> 11, n = gm & 2047;
      vTb[(((size_t)(b * 16 + h)) * 64 + d) * 2048 + n] = (bf16)Cs[r * 65 + d];
    }
  } else {
    bf16* dst = (which == 0) ? qb : kb;
    for (int t = tid; t < 128 * 64; t += 256) {
      int r = t >> 6, d = t & 63;
      int gm = mblk + r;
      int b = gm >> 11, n = gm & 2047;
      // llama RoPE: out[d] = x[d]*cos + (d<32 ? -x[d+32] : x[d-32])*sin
      float x = Cs[r * 65 + d];
      int f = d & 31;
      float inv = __expf(-(float)f * (9.210340371976184f / 32.0f)); // 10000^(-f/32)
      float s, c;
      __sincosf((float)n * inv, &s, &c);
      float partner = Cs[r * 65 + (d ^ 32)];
      float outv = x * c + ((d < 32) ? -partner : partner) * s;
      dst[(((size_t)(b * 16 + h)) * 2048 + n) * 64 + d] = (bf16)outv;
    }
  }
}

// ---------------------------------------------------------------------------
// Flash attention: grid (N/128, B*H), 256 threads (8 independent waves),
// each wave owns 16 q rows. K-frags and V-frags stream straight from global
// (K row-major [n][d] and V transposed [d][n] are both B-frag layouts).
// Only LDS use: per-wave P tile round-trip (wave-local, barrier-free).
// ---------------------------------------------------------------------------
__global__ __launch_bounds__(256) void k_flash_attn(const bf16* __restrict__ q,
                                                    const bf16* __restrict__ k,
                                                    const bf16* __restrict__ vT,
                                                    bf16* __restrict__ o) {
  __shared__ alignas(32) bf16 Ps[8][16][32];     // per-wave P tile, row-major

  int tid = threadIdx.x, lane = tid & 31, wave = tid >> 5;
  int bh = blockIdx.y;                            // b*16 + h
  int qrow = blockIdx.x * 128 + wave * 16;
  const size_t base = (size_t)bh * 2048 * 64;     // same flat size for [N,D] and [D,N]

  // Q A-frags (16 rows x 64 d) resident for the whole pass
  const bf16* qp = q + base + (size_t)qrow * 64;
  v16bf aq0 = load_a_frag(qp, 64, lane);          // d 0..31
  v16bf aq1 = load_a_frag(qp + 32, 64, lane);     // d 32..63

  v8f acc[4] = {};
  float mrow[8], lrow[8];
#pragma unroll
  for (int i = 0; i < 8; ++i) { mrow[i] = -1e30f; lrow[i] = 0.f; }
  const float scale = 0.125f;                     // 1/sqrt(64)

  for (int kt = 0; kt < 2048; kt += 32) {
    // S = Q K^T : K rows are d-contiguous == transposed-B layout
    const bf16* kp0 = k + base + (size_t)kt * 64;
    v16bf b00 = load_b_frag_t(kp0, 64, lane);               // cols 0-15,  d 0-31
    v16bf b01 = load_b_frag_t(kp0 + 32, 64, lane);          // cols 0-15,  d 32-63
    v16bf b10 = load_b_frag_t(kp0 + 16 * 64, 64, lane);     // cols 16-31, d 0-31
    v16bf b11 = load_b_frag_t(kp0 + 16 * 64 + 32, 64, lane);
    v8f s0 = {}, s1 = {};
    s0 = WMMA_BF16(aq0, b00, s0);
    s0 = WMMA_BF16(aq1, b01, s0);
    s1 = WMMA_BF16(aq0, b10, s1);
    s1 = WMMA_BF16(aq1, b11, s1);

    if (kt + 32 < 2048) {
      __builtin_prefetch((const void*)(k + base + (size_t)(kt + 32) * 64), 0, 3);
      __builtin_prefetch((const void*)(vT + base + (size_t)(kt + 32)), 0, 3);
    }

    // online softmax; rows live in 16-lane halves -> shfl_xor masks 1,2,4,8
    float corr[8];
#pragma unroll
    for (int vv = 0; vv < 8; ++vv) {
      float a = s0[vv] * scale, b2 = s1[vv] * scale;
      float mx = fmaxf(a, b2);
#pragma unroll
      for (int m2 = 1; m2 < 16; m2 <<= 1) mx = fmaxf(mx, __shfl_xor(mx, m2, 32));
      float mn = fmaxf(mrow[vv], mx);
      float p0 = __expf(a - mn), p1 = __expf(b2 - mn);
      float ps = p0 + p1;
#pragma unroll
      for (int m2 = 1; m2 < 16; m2 <<= 1) ps += __shfl_xor(ps, m2, 32);
      corr[vv] = __expf(mrow[vv] - mn);
      lrow[vv] = lrow[vv] * corr[vv] + ps;
      mrow[vv] = mn;
      // stash P as bf16 in this wave's private LDS tile (row-major 16x32);
      // LDS ops are in-order within a wave -> no barrier needed
      int r = vv + ((lane >> 4) << 3);
      int c = lane & 15;
      Ps[wave][r][c]      = (bf16)p0;
      Ps[wave][r][c + 16] = (bf16)p1;
    }
#pragma unroll
    for (int f = 0; f < 4; ++f)
#pragma unroll
      for (int vv = 0; vv < 8; ++vv) acc[f][vv] *= corr[vv];

    // O += P @ V : V transposed [d][n] is exactly the B-frag layout
    v16bf ap = load_a_frag(&Ps[wave][0][0], 32, lane);
#pragma unroll
    for (int f = 0; f < 4; ++f) {
      v16bf bv = load_b_frag_t(vT + base + (size_t)(f * 16) * 2048 + kt, 2048, lane);
      acc[f] = WMMA_BF16(ap, bv, acc[f]);
    }
  }

  // normalize and write [B,N,H*D] bf16 for the output projection
  int b = bh >> 4, h = bh & 15;
#pragma unroll
  for (int f = 0; f < 4; ++f)
#pragma unroll
    for (int vv = 0; vv < 8; ++vv) {
      int r = vv + ((lane >> 4) << 3);
      int n = qrow + r;
      int d = f * 16 + (lane & 15);
      float val = acc[f][vv] / lrow[vv];
      o[((size_t)(b * 2048 + n)) * 1024 + h * 64 + d] = (bf16)val;
    }
}

// ---------------------------------------------------------------------------
// GEMM2: out = attn @ w_out (pre-transposed), f32 result straight to d_out
// ---------------------------------------------------------------------------
__global__ __launch_bounds__(256) void k_gemm_out(const bf16* __restrict__ A,
                                                  const bf16* __restrict__ Wt,
                                                  float* __restrict__ C) {
  __shared__ alignas(32) bf16 As[2 * 128 * 32];

  const int K = 1024, NCOL = 1024;
  int tid = threadIdx.x, lane = tid & 31, wave = tid >> 5;
  int wm = wave & 3, wn = wave >> 2;
  int mblk = blockIdx.x * 128, nblk = blockIdx.y * 64;

  v8f acc[2][2] = {};
  gemm_mainloop_t(A, Wt, K, mblk, nblk, As, acc, tid, lane, wm, wn);

#pragma unroll
  for (int i = 0; i < 2; ++i)
#pragma unroll
    for (int j = 0; j < 2; ++j)
#pragma unroll
      for (int v = 0; v < 8; ++v) {
        int gm = mblk + wm * 32 + i * 16 + v + ((lane >> 4) << 3);
        int gc = nblk + wn * 32 + j * 16 + (lane & 15);
        C[(size_t)gm * NCOL + gc] = acc[i][j][v];
      }
}

// ---------------------------------------------------------------------------
// Launch
// ---------------------------------------------------------------------------
extern "C" void kernel_launch(void* const* d_in, const int* in_sizes, int n_in,
                              void* d_out, int out_size, void* d_ws, size_t ws_size,
                              hipStream_t stream) {
  (void)in_sizes; (void)n_in; (void)out_size; (void)ws_size;
  const float* x    = (const float*)d_in[0];   // [2,2048,1024]
  const float* wqkv = (const float*)d_in[1];   // [1024,3072]
  const float* wout = (const float*)d_in[2];   // [1024,1024]
  float* out = (float*)d_out;                  // [2,2048,1024]

  char* ws = (char*)d_ws;
  bf16* x_bf     = (bf16*)(ws);                       // 8 MB
  bf16* wqkvT_bf = (bf16*)(ws + ((size_t)8  << 20));  // 6 MB, [3072][1024]
  bf16* woutT_bf = (bf16*)(ws + ((size_t)14 << 20));  // 2 MB, [1024][1024]
  bf16* q_bf     = (bf16*)(ws + ((size_t)16 << 20));  // 8 MB, [B,H,N,D]
  bf16* k_bf     = (bf16*)(ws + ((size_t)24 << 20));  // 8 MB, [B,H,N,D]
  bf16* vT_bf    = (bf16*)(ws + ((size_t)32 << 20));  // 8 MB, [B,H,D,N]
  bf16* attn_bf  = (bf16*)(ws + ((size_t)40 << 20));  // 8 MB, [B,N,H*D]

  const int nx = 2 * 2048 * 1024;
  k_f32_to_bf16<<<nx / 1024, 256, 0, stream>>>(x, x_bf, nx);
  k_transpose_bf16<<<dim3(32, 96), 256, 0, stream>>>(wqkv, wqkvT_bf, 1024, 3072);
  k_transpose_bf16<<<dim3(32, 32), 256, 0, stream>>>(wout, woutT_bf, 1024, 1024);

  k_gemm_qkv_rope<<<dim3(32, 48), 256, 0, stream>>>(x_bf, wqkvT_bf, q_bf, k_bf, vT_bf);
  k_flash_attn<<<dim3(16, 32), 256, 0, stream>>>(q_bf, k_bf, vT_bf, attn_bf);
  k_gemm_out<<<dim3(32, 16), 256, 0, stream>>>(attn_bf, woutT_bf, out);
}
